// TimeDecoder_67903432950244
// MI455X (gfx1250) — matrix operand order repaired and checked by reading, not compiled, not run on hardware
//
#include <hip/hip_runtime.h>

#define Bsz 128
#define Tsz 512
#define Fsz 16
#define Hsz 32
#define Ksz 5
#define NEGS 0.01f

typedef __attribute__((ext_vector_type(16))) _Float16 v16h;
typedef __attribute__((ext_vector_type(8)))  _Float16 v8h;
typedef __attribute__((ext_vector_type(8)))  float    v8f;

static constexpr long BT = (long)Bsz * Tsz; // 65536

// ---- CDNA5 WMMA fragment helpers (wave32; contiguous K = lh*16 + e in both A and B) ----

__device__ __forceinline__ int kA(int e, int lh) {
  // f16 A 16x32: K = e + lh*16
  return e + lh * 16;
}

// A fragment: one contiguous 32B run per lane (K = lh*16 + e) -> 2 wide loads
__device__ __forceinline__ v16h load_a2(const _Float16* rowbase, int lh) {
  v8h lo = *(const v8h*)(rowbase + lh * 16);
  v8h hi = *(const v8h*)(rowbase + lh * 16 + 8);
  return __builtin_shufflevector(lo, hi, 0, 1, 2, 3, 4, 5, 6, 7, 8, 9, 10, 11, 12, 13, 14, 15);
}

// B fragment from K-contiguous storage: one 32B run per lane -> 2 wide loads
__device__ __forceinline__ v16h load_b2(const _Float16* base, int lh) {
  v8h lo = *(const v8h*)(base + lh * 16);
  v8h hi = *(const v8h*)(base + lh * 16 + 8);
  return __builtin_shufflevector(lo, hi, 0, 1, 2, 3, 4, 5, 6, 7, 8, 9, 10, 11, 12, 13, 14, 15);
}

__device__ __forceinline__ v8f wmma_f16(v16h a, v16h b, v8f c) {
  return __builtin_amdgcn_wmma_f32_16x16x32_f16(false, a, false, b, (short)0, c, false, false);
}

// ---- fast activations: CDNA5 has native v_tanh_f32 (TRANS32 op); fall back to
// branch-free exp+rcp (clamped so no inf/inf) if the builtin is unavailable ----
__device__ __forceinline__ float fast_rcp(float x) {
#if __has_builtin(__builtin_amdgcn_rcpf)
  return __builtin_amdgcn_rcpf(x);
#else
  return 1.f / x;
#endif
}

__device__ __forceinline__ float tanh_(float x) {
#if __has_builtin(__builtin_amdgcn_tanhf)
  return __builtin_amdgcn_tanhf(x);
#else
  float xc = fminf(fmaxf(x, -15.f), 15.f);
  float e = __expf(-2.f * xc);
  return (1.f - e) * fast_rcp(1.f + e);
#endif
}

__device__ __forceinline__ float sigmoid_(float x) {
#if __has_builtin(__builtin_amdgcn_tanhf)
  return __builtin_fmaf(0.5f, __builtin_amdgcn_tanhf(0.5f * x), 0.5f);
#else
  float xc = fminf(fmaxf(x, -30.f), 30.f);
  float e = __expf(-xc);
  return fast_rcp(1.f + e);
#endif
}

__device__ __forceinline__ float leaky_(float v) { return v >= 0.f ? v : NEGS * v; }

// ---- Kernel 1: encoder input projection, transposed: xpT[t][48][B] ----
__global__ void enc_xproj_kernel(const float* __restrict__ x, const float* __restrict__ Wih,
                                 const float* __restrict__ bih, float* __restrict__ xpT) {
  long i = (long)blockIdx.x * blockDim.x + threadIdx.x;
  if (i >= BT * 48) return;
  int b = (int)(i & (Bsz - 1));
  long rest = i >> 7;
  int g = (int)(rest % 48);
  int t = (int)(rest / 48);
  float x0 = x[(long)b * 2 * Tsz + t];
  float x1 = x[(long)b * 2 * Tsz + Tsz + t];
  xpT[i] = x0 * Wih[g * 2] + x1 * Wih[g * 2 + 1] + bih[g];
}

// ---- Kernel 1b: swizzle conv weights into WMMA A-fragment order (f16) ----
__global__ void wswz_kernel(const float* __restrict__ w3, const float* __restrict__ w2,
                            _Float16* __restrict__ wf3, _Float16* __restrict__ wf2) {
  const long n3 = (long)Fsz * 5 * 4 * 32 * 16;      // 163840
  const long n2 = (long)Fsz * 5 * 2 * 8 * 32 * 16;  // 655360
  long i = (long)blockIdx.x * blockDim.x + threadIdx.x;
  if (i < n3) {
    int e = (int)(i & 15);
    int lane = (int)((i >> 4) & 31);
    int cot = (int)((i >> 9) & 3);
    long rec = i >> 11;               // f*5 + k
    int k = (int)(rec % 5), f = (int)(rec / 5);
    int ml = lane & 15, lh = lane >> 4;
    int co = cot * 16 + ml;
    int ci = kA(e, lh);
    wf3[i] = (_Float16)w3[(((long)f * 32 + ci) * 64 + co) * Ksz + (Ksz - 1 - k)];
  } else if (i < n3 + n2) {
    long j = i - n3;
    int e = (int)(j & 15);
    int lane = (int)((j >> 4) & 31);
    int cot = (int)((j >> 9) & 7);
    int kc = (int)((j >> 12) & 1);
    long rec = j >> 13;               // f*5 + k
    int k = (int)(rec % 5), f = (int)(rec / 5);
    int ml = lane & 15, lh = lane >> 4;
    int co = cot * 16 + ml;
    int ci = kc * 32 + kA(e, lh);
    wf2[j] = (_Float16)w2[(((long)f * 64 + ci) * 128 + co) * Ksz + (Ksz - 1 - k)];
  }
}

// ---- Kernel 2: encoder GRU scan (H=16 padded to K=32); gates computed on fragments ----
__global__ void __launch_bounds__(256) enc_scan_kernel(const float* __restrict__ Whh,
                                                       const float* __restrict__ bhh,
                                                       const float* __restrict__ xpT,
                                                       float* __restrict__ feat) {
  __shared__ alignas(32) _Float16 hF16[Bsz * 32];   // K-padded hidden (cols 16..31 stay 0)

  int tid = threadIdx.x;
  for (int i = tid; i < Bsz * 32; i += 256) hF16[i] = (_Float16)0.f;
  __syncthreads();

  int lane = tid & 31, wv = tid >> 5;
  int lh = lane >> 4, ml = lane & 15, nl = lane & 15;
  int b0 = wv * 16;

  v16h BH[3];
#pragma unroll
  for (int nt = 0; nt < 3; ++nt) {
    v16h bb;
#pragma unroll
    for (int e = 0; e < 16; ++e) {
      int k = e + lh * 16;
      bb[e] = (k < 16) ? (_Float16)Whh[(nt * 16 + nl) * 16 + k] : (_Float16)0.f;
    }
    BH[nt] = bb;
  }
  int ch = nl;
  float bh_r = bhh[ch], bh_z = bhh[16 + ch], bh_n = bhh[32 + ch];
  float hreg[8];
#pragma unroll
  for (int j = 0; j < 8; ++j) hreg[j] = 0.f;

  const _Float16* hA = hF16 + (b0 + ml) * 32;       // A-row base (per lane)
  _Float16* hS = hF16 + (b0 + 8 * lh) * 32 + ch;    // h store base (per lane)
  const float* xq = xpT + (long)ch * Bsz + b0 + 8 * lh;  // xp base, +48*Bsz per step
  float* fs = feat + (long)ch * BT + (long)(b0 + 8 * lh) * Tsz;  // feat base, +1 per step

  for (int t = 0; t < Tsz; ++t) {
    v16h a = load_a2(hA, lh);
    v8f cr = wmma_f16(a, BH[0], (v8f){});
    v8f cz = wmma_f16(a, BH[1], (v8f){});
    v8f cn = wmma_f16(a, BH[2], (v8f){});
#pragma unroll
    for (int j = 0; j < 8; ++j) {
      float r = sigmoid_(xq[j * Bsz * 0 + j] + cr[j] + bh_r);   // xq[j]
      float z = sigmoid_(xq[16 * Bsz + j] + cz[j] + bh_z);
      float n = tanh_(xq[32 * Bsz + j] + r * (cn[j] + bh_n));
      float h = (1.f - z) * n + z * hreg[j];
      hreg[j] = h;
      hS[j * 32] = (_Float16)h;
      fs[(long)j * Tsz] = h;
    }
    xq += 48 * Bsz;
    fs += 1;
  }
}

// ---- Kernel 3: g1 biGRU scan (I=1, H=32); per-wave recurrence with h staged in LDS ----
__global__ void __launch_bounds__(256) g1_scan_kernel(
    const float* __restrict__ Wih_f, const float* __restrict__ Whh_f,
    const float* __restrict__ bih_f, const float* __restrict__ bhh_f,
    const float* __restrict__ Wih_b, const float* __restrict__ Whh_b,
    const float* __restrict__ bih_b, const float* __restrict__ bhh_b,
    const float* __restrict__ feat, _Float16* __restrict__ u) {
  int f = blockIdx.x, dir = blockIdx.y;
  const float* Wih = dir ? Wih_b : Wih_f;
  const float* Whh = dir ? Whh_b : Whh_f;
  const float* bih = dir ? bih_b : bih_f;
  const float* bhh = dir ? bhh_b : bhh_f;

  __shared__ alignas(32) _Float16 hF16[Bsz * 32];
  int tid = threadIdx.x;
  for (int i = tid; i < Bsz * 32; i += 256) hF16[i] = (_Float16)0.f;
  __syncthreads();

  int lane = tid & 31, wv = tid >> 5;
  int lh = lane >> 4, ml = lane & 15, nl = lane & 15;
  int b0 = wv * 16;

  v16h BH[6];
#pragma unroll
  for (int nt = 0; nt < 6; ++nt) {
    v16h bb;
#pragma unroll
    for (int e = 0; e < 16; ++e)
      bb[e] = (_Float16)Whh[((long)f * 96 + nt * 16 + nl) * 32 + e + lh * 16];
    BH[nt] = bb;
  }
  int c1 = nl, c2 = nl + 16;
  const float* wq = Wih + (long)f * 96;
  const float* biq = bih + (long)f * 96;
  const float* bhq = bhh + (long)f * 96;
  float wi_r1 = wq[c1], wi_z1 = wq[32 + c1], wi_n1 = wq[64 + c1];
  float wi_r2 = wq[c2], wi_z2 = wq[32 + c2], wi_n2 = wq[64 + c2];
  float bi_r1 = biq[c1], bi_z1 = biq[32 + c1], bi_n1 = biq[64 + c1];
  float bi_r2 = biq[c2], bi_z2 = biq[32 + c2], bi_n2 = biq[64 + c2];
  float bh_r1 = bhq[c1], bh_z1 = bhq[32 + c1], bh_n1 = bhq[64 + c1];
  float bh_r2 = bhq[c2], bh_z2 = bhq[32 + c2], bh_n2 = bhq[64 + c2];

  float h1[8], h2r[8];
#pragma unroll
  for (int j = 0; j < 8; ++j) { h1[j] = 0.f; h2r[j] = 0.f; }

  int t0 = dir ? (Tsz - 1) : 0;
  long dt = dir ? -1 : 1;
  const _Float16* hA = hF16 + (b0 + ml) * 32;
  _Float16* hS = hF16 + (b0 + 8 * lh) * 32;
  const float* fp = feat + (long)f * BT + (long)(b0 + 8 * lh) * Tsz + t0;
  _Float16* up = u + ((long)f * BT + (long)(b0 + 8 * lh) * Tsz + t0) * 64 + dir * 32;

  for (int s = 0; s < Tsz; ++s) {
    v16h a = load_a2(hA, lh);
    v8f acc[6];
#pragma unroll
    for (int nt = 0; nt < 6; ++nt) acc[nt] = wmma_f16(a, BH[nt], (v8f){});
#pragma unroll
    for (int j = 0; j < 8; ++j) {
      float xi = fp[(long)j * Tsz];
      float r = sigmoid_(xi * wi_r1 + bi_r1 + acc[0][j] + bh_r1);
      float z = sigmoid_(xi * wi_z1 + bi_z1 + acc[2][j] + bh_z1);
      float n = tanh_(xi * wi_n1 + bi_n1 + r * (acc[4][j] + bh_n1));
      h1[j] = (1.f - z) * n + z * h1[j];
      float r2 = sigmoid_(xi * wi_r2 + bi_r2 + acc[1][j] + bh_r2);
      float z2 = sigmoid_(xi * wi_z2 + bi_z2 + acc[3][j] + bh_z2);
      float n2 = tanh_(xi * wi_n2 + bi_n2 + r2 * (acc[5][j] + bh_n2));
      h2r[j] = (1.f - z2) * n2 + z2 * h2r[j];
      hS[j * 32 + c1] = (_Float16)h1[j];
      hS[j * 32 + c2] = (_Float16)h2r[j];
      up[(long)j * Tsz * 64 + c1] = (_Float16)h1[j];
      up[(long)j * Tsz * 64 + c2] = (_Float16)h2r[j];
    }
    fp += dt;
    up += dt * 64;
  }
}

// ---- Kernel 4: g2 GRU scan (I=64, H=32); input proj fused as extra WMMA K-chunks ----
__global__ void __launch_bounds__(256) g2_scan_kernel(
    const float* __restrict__ Wih, const float* __restrict__ Whh,
    const float* __restrict__ bih, const float* __restrict__ bhh,
    const _Float16* __restrict__ u, _Float16* __restrict__ h2) {
  int f = blockIdx.x;
  __shared__ alignas(32) _Float16 hF16[Bsz * 32];
  __shared__ alignas(32) _Float16 WxL[12 * 32 * 16];  // Wih^T B fragments, per-lane records

  int tid = threadIdx.x;
  for (int i = tid; i < Bsz * 32; i += 256) hF16[i] = (_Float16)0.f;
  for (int i = tid; i < 12 * 32 * 16; i += 256) {
    int e = i & 15, ln = (i >> 4) & 31, fr = i >> 9;  // fr = nt*2 + kc
    int nt = fr >> 1, kc = fr & 1;
    int nn = ln & 15, hh = ln >> 4;
    WxL[i] = (_Float16)Wih[((long)f * 96 + nt * 16 + nn) * 64 + kc * 32 + e + hh * 16];
  }
  __syncthreads();

  int lane = tid & 31, wv = tid >> 5;
  int lh = lane >> 4, ml = lane & 15, nl = lane & 15;
  int b0 = wv * 16;

  v16h BH[6];
#pragma unroll
  for (int nt = 0; nt < 6; ++nt) {
    v16h bb;
#pragma unroll
    for (int e = 0; e < 16; ++e)
      bb[e] = (_Float16)Whh[((long)f * 96 + nt * 16 + nl) * 32 + e + lh * 16];
    BH[nt] = bb;
  }
  int c1 = nl, c2 = nl + 16;
  const float* biq = bih + (long)f * 96;
  const float* bhq = bhh + (long)f * 96;
  float bs_r1 = biq[c1] + bhq[c1], bs_z1 = biq[32 + c1] + bhq[32 + c1];
  float bs_r2 = biq[c2] + bhq[c2], bs_z2 = biq[32 + c2] + bhq[32 + c2];
  float bi_n1 = biq[64 + c1], bh_n1 = bhq[64 + c1];
  float bi_n2 = biq[64 + c2], bh_n2 = bhq[64 + c2];

  float h1[8], h2v[8];
#pragma unroll
  for (int j = 0; j < 8; ++j) { h1[j] = 0.f; h2v[j] = 0.f; }

  const _Float16* hA = hF16 + (b0 + ml) * 32;
  _Float16* hS = hF16 + (b0 + 8 * lh) * 32;
  const _Float16* up = u + ((long)f * BT + (long)(b0 + ml) * Tsz) * 64;   // +64 per step
  _Float16* o1 = h2 + ((long)f * 32 + c1) * BT + (long)(b0 + 8 * lh) * Tsz;  // +1 per step
  _Float16* o2 = h2 + ((long)f * 32 + c2) * BT + (long)(b0 + 8 * lh) * Tsz;

  for (int t = 0; t < Tsz; ++t) {
    __builtin_prefetch(up + 64, 0, 1);
    v16h aU0 = load_a2(up, lh);
    v16h aU1 = load_a2(up + 32, lh);
    v16h aH  = load_a2(hA, lh);

    v8f comb[4];  // r/z gates: xproj + hproj chained into one accumulator
#pragma unroll
    for (int nt = 0; nt < 4; ++nt) {
      v8f c = wmma_f16(aH, BH[nt], (v8f){});
      c = wmma_f16(aU0, *(const v16h*)(WxL + ((nt * 2 + 0) * 32 + lane) * 16), c);
      c = wmma_f16(aU1, *(const v16h*)(WxL + ((nt * 2 + 1) * 32 + lane) * 16), c);
      comb[nt] = c;
    }
    v8f cxn[2], chn[2];  // n gate needs xproj / hproj separate
#pragma unroll
    for (int q = 0; q < 2; ++q) {
      int nt = 4 + q;
      v8f cx = wmma_f16(aU0, *(const v16h*)(WxL + ((nt * 2 + 0) * 32 + lane) * 16), (v8f){});
      cx = wmma_f16(aU1, *(const v16h*)(WxL + ((nt * 2 + 1) * 32 + lane) * 16), cx);
      cxn[q] = cx;
      chn[q] = wmma_f16(aH, BH[nt], (v8f){});
    }
#pragma unroll
    for (int j = 0; j < 8; ++j) {
      float r = sigmoid_(comb[0][j] + bs_r1);
      float z = sigmoid_(comb[2][j] + bs_z1);
      float n = tanh_(cxn[0][j] + bi_n1 + r * (chn[0][j] + bh_n1));
      h1[j] = (1.f - z) * n + z * h1[j];
      float r2 = sigmoid_(comb[1][j] + bs_r2);
      float z2 = sigmoid_(comb[3][j] + bs_z2);
      float n2 = tanh_(cxn[1][j] + bi_n2 + r2 * (chn[1][j] + bh_n2));
      h2v[j] = (1.f - z2) * n2 + z2 * h2v[j];
      hS[j * 32 + c1] = (_Float16)h1[j];
      hS[j * 32 + c2] = (_Float16)h2v[j];
      o1[(long)j * Tsz] = (_Float16)h1[j];
      o2[(long)j * Tsz] = (_Float16)h2v[j];
    }
    up += 64;
    o1 += 1;
    o2 += 1;
  }
}

// ---- Kernel 5: fused convT3 -> convT2 -> convT1 per 64-column tile, column-major LDS ----
__global__ void __launch_bounds__(256) conv_fused_kernel(
    const _Float16* __restrict__ wf3, const float* __restrict__ b3,
    const _Float16* __restrict__ wf2, const float* __restrict__ b2,
    const float* __restrict__ w1, const float* __restrict__ b1,
    const _Float16* __restrict__ h2, float* __restrict__ out) {
  int f = blockIdx.y;
  int b = blockIdx.x >> 3;
  int tbase = (blockIdx.x & 7) * 64;

  __shared__ alignas(32) _Float16 X0t[88 * 32];   // [col][ci], col c -> t = tbase - 6 + c
  __shared__ alignas(32) _Float16 X1t[88 * 64];   // [col][ci], col n -> t = tbase - 4 + n
  __shared__ alignas(32) _Float16 X2t[80 * 128];  // [col][ci], col m -> t = tbase - 2 + m
  __shared__ float wf1L[5 * 128];

  int tid = threadIdx.x;
  for (int i = tid; i < 32 * 88; i += 256) {
    int ci = i / 88, c = i % 88;
    int t = tbase - 6 + c;
    X0t[c * 32 + ci] = (t >= 0 && t < Tsz)
        ? h2[((long)f * 32 + ci) * BT + (long)b * Tsz + t] : (_Float16)0.f;
  }
  for (int i = tid; i < 5 * 128; i += 256) {
    int k = i / 128, ci = i % 128;
    wf1L[i] = w1[((long)f * 128 + ci) * Ksz + (Ksz - 1 - k)];
  }
  __syncthreads();

  int lane = tid & 31, wv = tid >> 5;
  int lh = lane >> 4, nl = lane & 15;

  // stage 1: 32 -> 64
  for (int tile = wv; tile < 20; tile += 8) {
    int cot = tile / 5, n0 = (tile % 5) * 16;
    v8f c = {};
#pragma unroll
    for (int k = 0; k < Ksz; ++k) {
      v16h a = *(const v16h*)(wf3 + ((((long)f * 5 + k) * 4 + cot) * 32 + lane) * 16);
      v16h bb = load_b2(X0t + (n0 + k + nl) * 32, lh);
      c = wmma_f16(a, bb, c);
    }
    int co0 = cot * 16;
    v8h s;
#pragma unroll
    for (int r = 0; r < 8; ++r)
      s[r] = (_Float16)leaky_(c[r] + b3[f * 64 + co0 + 8 * lh + r]);
    *(v8h*)(X1t + (n0 + nl) * 64 + co0 + 8 * lh) = s;
  }
  __syncthreads();
  for (int n = tid; n < 88; n += 256) {   // conv zero-padding in t + uninit tail
    int t = tbase - 4 + n;
    if (n >= 80 || t < 0 || t >= Tsz) {
      v8h z = {};
#pragma unroll
      for (int q = 0; q < 8; ++q) *(v8h*)(X1t + n * 64 + q * 8) = z;
    }
  }
  __syncthreads();

  // stage 2: 64 -> 128
  for (int tile = wv; tile < 40; tile += 8) {
    int cot = tile / 5, n0 = (tile % 5) * 16;
    v8f c = {};
#pragma unroll
    for (int k = 0; k < Ksz; ++k) {
#pragma unroll
      for (int kc = 0; kc < 2; ++kc) {
        v16h a = *(const v16h*)(wf2 + (((((long)f * 5 + k) * 2 + kc) * 8 + cot) * 32 + lane) * 16);
        v16h bb = load_b2(X1t + (n0 + k + nl) * 64 + kc * 32, lh);
        c = wmma_f16(a, bb, c);
      }
    }
    int co0 = cot * 16;
    v8h s;
#pragma unroll
    for (int r = 0; r < 8; ++r)
      s[r] = (_Float16)leaky_(c[r] + b2[f * 128 + co0 + 8 * lh + r]);
    *(v8h*)(X2t + (n0 + nl) * 128 + co0 + 8 * lh) = s;
  }
  __syncthreads();
  for (int m = tid; m < 80; m += 256) {
    int t = tbase - 2 + m;
    if (t < 0 || t >= Tsz) {
      v8h z = {};
#pragma unroll
      for (int q = 0; q < 16; ++q) *(v8h*)(X2t + m * 128 + q * 8) = z;
    }
  }
  __syncthreads();

  // stage 3: 128 -> 1, vectorized scalar reduction
  if (tid < 64) {
    int j = tid;
    float acc = b1[f];
#pragma unroll
    for (int k = 0; k < Ksz; ++k) {
      const _Float16* xr = X2t + (j + k) * 128;
      for (int c8 = 0; c8 < 16; ++c8) {
        v8h xv = *(const v8h*)(xr + c8 * 8);
#pragma unroll
        for (int e = 0; e < 8; ++e) acc += (float)xv[e] * wf1L[k * 128 + c8 * 8 + e];
      }
    }
    out[((long)b * Tsz + tbase + j) * Fsz + f] = leaky_(acc);
  }
}

extern "C" void kernel_launch(void* const* d_in, const int* in_sizes, int n_in,
                              void* d_out, int out_size, void* d_ws, size_t ws_size,
                              hipStream_t stream) {
  (void)in_sizes; (void)n_in; (void)out_size; (void)ws_size;
  const float* x        = (const float*)d_in[0];
  const float* enc_Wih  = (const float*)d_in[1];
  const float* enc_Whh  = (const float*)d_in[2];
  const float* enc_bih  = (const float*)d_in[3];
  const float* enc_bhh  = (const float*)d_in[4];
  const float* g1_Wih_f = (const float*)d_in[5];
  const float* g1_Whh_f = (const float*)d_in[6];
  const float* g1_bih_f = (const float*)d_in[7];
  const float* g1_bhh_f = (const float*)d_in[8];
  const float* g1_Wih_b = (const float*)d_in[9];
  const float* g1_Whh_b = (const float*)d_in[10];
  const float* g1_bih_b = (const float*)d_in[11];
  const float* g1_bhh_b = (const float*)d_in[12];
  const float* g2_Wih   = (const float*)d_in[13];
  const float* g2_Whh   = (const float*)d_in[14];
  const float* g2_bih   = (const float*)d_in[15];
  const float* g2_bhh   = (const float*)d_in[16];
  const float* w3       = (const float*)d_in[17];
  const float* b3       = (const float*)d_in[18];
  const float* w2       = (const float*)d_in[19];
  const float* b2       = (const float*)d_in[20];
  const float* w1       = (const float*)d_in[21];
  const float* b1       = (const float*)d_in[22];

  char* ws = (char*)d_ws;
  float* xpT       = (float*)ws;                                       // BT*48 f32
  float* feat      = (float*)(ws + (size_t)BT * 48 * 4);               // F*BT f32
  _Float16* u      = (_Float16*)((char*)feat + (size_t)Fsz * BT * 4);  // F*BT*64 f16
  _Float16* h2buf  = (_Float16*)((char*)u + (size_t)Fsz * BT * 64 * 2);      // F*32*BT f16
  _Float16* wf3    = (_Float16*)((char*)h2buf + (size_t)Fsz * 32 * BT * 2);  // 163840 f16
  _Float16* wf2    = wf3 + (size_t)Fsz * 5 * 4 * 32 * 16;                    // 655360 f16
  float* out = (float*)d_out;

  enc_xproj_kernel<<<(unsigned)((BT * 48 + 255) / 256), 256, 0, stream>>>(x, enc_Wih, enc_bih, xpT);
  wswz_kernel<<<(unsigned)((Fsz * 5 * 4 * 512 + Fsz * 5 * 16 * 512 + 255) / 256), 256, 0, stream>>>(
      w3, w2, wf3, wf2);
  enc_scan_kernel<<<1, 256, 0, stream>>>(enc_Whh, enc_bhh, xpT, feat);
  g1_scan_kernel<<<dim3(Fsz, 2), 256, 0, stream>>>(g1_Wih_f, g1_Whh_f, g1_bih_f, g1_bhh_f,
                                                   g1_Wih_b, g1_Whh_b, g1_bih_b, g1_bhh_b,
                                                   feat, u);
  g2_scan_kernel<<<Fsz, 256, 0, stream>>>(g2_Wih, g2_Whh, g2_bih, g2_bhh, u, h2buf);
  conv_fused_kernel<<<dim3((unsigned)(BT / 64), Fsz), 256, 0, stream>>>(wf3, b3, wf2, b2,
                                                                        w1, b1, h2buf, out);
}
